// FACE_33672543601387
// MI455X (gfx1250) — matrix operand order repaired and checked by reading, not compiled, not run on hardware
//
#include <hip/hip_runtime.h>

typedef __bf16 bf16_t;
typedef __bf16 v16bf __attribute__((ext_vector_type(16)));
typedef __bf16 v8bf  __attribute__((ext_vector_type(8)));
typedef float  v8f   __attribute__((ext_vector_type(8)));

#define NTOK   16384     // B * WORD_NUM
#define WD     256       // WORD_DIM
#define NCODE  16384     // V
#define LOSS_OFF 131072
#define TOK_OFF  131073
#define TTQ_OFF  147457

// ---------------------------------------------------------------------------
// Generic WMMA GEMM:  C[M x N] (f32) = A[M x K] (f32->bf16) * W^T + bias
// W stored (N x K) row-major (weight convention), or (K x N) if wTrans.
// Block: 256 threads = 8 waves; block tile 128(M) x 64(N); wave tile 32x32.
// Double-buffered LDS staging: one barrier per 32-K step, global loads of
// tile t+1 overlap the WMMAs of tile t.
// ---------------------------------------------------------------------------
#define LDSA_STR 48   // 96B row stride: 16B/32B aligned fragment loads
#define LDSB_STR 48

__device__ inline v16bf make_afrag(const bf16_t* base /* row base */, int h) {
    // A 16x32 bf16 fragment: lane half h holds K in [h*16,h*16+16) (contiguous, matches B)
    v8bf lo = *(const v8bf*)(base + h * 16);
    v8bf hi = *(const v8bf*)(base + h * 16 + 8);
    return __builtin_shufflevector(lo, hi, 0,1,2,3,4,5,6,7,8,9,10,11,12,13,14,15);
}

__global__ __launch_bounds__(256)
void gemm_wmma(const float* __restrict__ A, int lda, long long aBatch,
               const float* __restrict__ W, int ldw, long long wBatch, int wTrans,
               const float* __restrict__ bias, long long bBatch,
               float* __restrict__ C, int ldc, long long cBatch,
               int M, int N, int K, int relu)
{
    (void)M; (void)N;
    __shared__ bf16_t sA[2][128 * LDSA_STR];
    __shared__ bf16_t sB[2][64 * LDSB_STR];

    const int z = blockIdx.z;
    A += (long long)z * aBatch;
    W += (long long)z * wBatch;
    C += (long long)z * cBatch;
    const float* bz = bias ? bias + (long long)z * bBatch : nullptr;

    const int tid = threadIdx.x;
    const int lane = tid & 31, wave = tid >> 5;
    const int wm = (wave >> 1) * 32;   // 0,32,64,96
    const int wn = (wave & 1) * 32;    // 0,32
    const int blockM = blockIdx.y * 128;
    const int blockN = blockIdx.x * 64;

    v8f acc[2][2] = {};

    const int arow = tid >> 1, achunk = (tid & 1) * 16;         // A tile 128x32
    const int brow = (tid & 127) >> 1, bchunk = (tid & 1) * 16; // W tile 64x32

    const int rl = lane & 15;
    const int h  = lane >> 4;

    auto stage = [&](int k0, int buf) {
        {   // stage A (f32 -> bf16)
            const float* src = A + (long long)(blockM + arow) * lda + k0 + achunk;
            bf16_t* dst = sA[buf] + arow * LDSA_STR + achunk;
            #pragma unroll
            for (int i = 0; i < 16; ++i) dst[i] = (bf16_t)src[i];
        }
        if (tid < 128) { // stage W
            bf16_t* dst = sB[buf] + brow * LDSB_STR + bchunk;
            if (!wTrans) {
                const float* src = W + (long long)(blockN + brow) * ldw + k0 + bchunk;
                #pragma unroll
                for (int i = 0; i < 16; ++i) dst[i] = (bf16_t)src[i];
            } else { // W is (K x N); element (n=brow, k) at W[k*ldw + n]
                const float* src = W + (long long)(k0 + bchunk) * ldw + (blockN + brow);
                #pragma unroll
                for (int i = 0; i < 16; ++i) dst[i] = (bf16_t)src[(long long)i * ldw];
            }
        }
    };

    const int nT = K / 32;
    stage(0, 0);
    __syncthreads();

    for (int t = 0; t < nT; ++t) {
        const int cur = t & 1, nxt = cur ^ 1;
        if (t + 1 < nT) stage((t + 1) * 32, nxt);   // overlaps WMMAs below

        v16bf aF0 = make_afrag(sA[cur] + (wm +      rl) * LDSA_STR, h);
        v16bf aF1 = make_afrag(sA[cur] + (wm + 16 + rl) * LDSA_STR, h);
        v16bf bF0 = *(const v16bf*)&sB[cur][(wn +      rl) * LDSB_STR + h * 16];
        v16bf bF1 = *(const v16bf*)&sB[cur][(wn + 16 + rl) * LDSB_STR + h * 16];

        acc[0][0] = __builtin_amdgcn_wmma_f32_16x16x32_bf16(false, aF0, false, bF0, (short)0, acc[0][0], false, false);
        acc[0][1] = __builtin_amdgcn_wmma_f32_16x16x32_bf16(false, aF0, false, bF1, (short)0, acc[0][1], false, false);
        acc[1][0] = __builtin_amdgcn_wmma_f32_16x16x32_bf16(false, aF1, false, bF0, (short)0, acc[1][0], false, false);
        acc[1][1] = __builtin_amdgcn_wmma_f32_16x16x32_bf16(false, aF1, false, bF1, (short)0, acc[1][1], false, false);
        __syncthreads();   // nxt fully staged AND cur fully consumed
    }

    #pragma unroll
    for (int i = 0; i < 2; ++i)
    #pragma unroll
    for (int j = 0; j < 2; ++j) {
        const int n = blockN + wn + j * 16 + rl;
        const float bv = bz ? bz[n] : 0.0f;
        #pragma unroll
        for (int r = 0; r < 8; ++r) {
            const int m = blockM + wm + i * 16 + r + 8 * h;
            float v = acc[i][j][r] + bv;
            if (relu) v = fmaxf(v, 0.0f);
            C[(long long)m * ldc + n] = v;
        }
    }
}

// ---------------------------------------------------------------------------
// Attention over 8 tokens of dim 256, one block per batch element
// ---------------------------------------------------------------------------
__global__ __launch_bounds__(256)
void attn_kernel(const float* __restrict__ qkv, float* __restrict__ out)
{
    __shared__ float q[8][256], k[8][256], vv[8][256];
    __shared__ float p[8][8];
    const long long b = blockIdx.x;
    const float* base = qkv + b * 8 * 768;
    const int tid = threadIdx.x;

    for (int e = tid; e < 8 * 768; e += 256) {
        const int w = e / 768, c = e % 768;
        const float val = base[w * 768 + c];
        if (c < 256) q[w][c] = val;
        else if (c < 512) k[w][c - 256] = val;
        else vv[w][c - 512] = val;
    }
    __syncthreads();

    if (tid < 64) {
        const int i = tid >> 3, j = tid & 7;
        float acc = 0.0f;
        for (int d = 0; d < 256; ++d) acc += q[i][d] * k[j][d];
        p[i][j] = acc * 0.0625f;            // 1/sqrt(256)
    }
    __syncthreads();
    if (tid < 8) {
        float mx = -3.0e38f;
        #pragma unroll
        for (int j = 0; j < 8; ++j) mx = fmaxf(mx, p[tid][j]);
        float ex[8]; float sum = 0.0f;
        #pragma unroll
        for (int j = 0; j < 8; ++j) { ex[j] = __expf(p[tid][j] - mx); sum += ex[j]; }
        const float inv = 1.0f / sum;
        #pragma unroll
        for (int j = 0; j < 8; ++j) p[tid][j] = ex[j] * inv;
    }
    __syncthreads();
    {
        const int d = tid;
        #pragma unroll
        for (int i = 0; i < 8; ++i) {
            float acc = 0.0f;
            #pragma unroll
            for (int j = 0; j < 8; ++j) acc += p[i][j] * vv[j][d];
            out[(b * 8 + i) * 256 + d] = acc;
        }
    }
}

// ---------------------------------------------------------------------------
// LayerNorm(out = LN(x + a) * g + b), one block (256 thr) per row of 256
// ---------------------------------------------------------------------------
__global__ __launch_bounds__(256)
void ln_kernel(const float* __restrict__ X, const float* __restrict__ A,
               const float* __restrict__ g, const float* __restrict__ b,
               float* __restrict__ out)
{
    __shared__ float red[256];
    __shared__ float sm, sv;
    const long long row = blockIdx.x;
    const int t = threadIdx.x;
    const long long idx = row * 256 + t;
    const float v = X[idx] + A[idx];

    red[t] = v; __syncthreads();
    for (int s = 128; s > 0; s >>= 1) { if (t < s) red[t] += red[t + s]; __syncthreads(); }
    if (t == 0) sm = red[0] * (1.0f / 256.0f);
    __syncthreads();
    const float d = v - sm;
    red[t] = d * d; __syncthreads();
    for (int s = 128; s > 0; s >>= 1) { if (t < s) red[t] += red[t + s]; __syncthreads(); }
    if (t == 0) sv = red[0] * (1.0f / 256.0f);
    __syncthreads();
    out[idx] = d * rsqrtf(sv + 1e-5f) * g[t] + b[t];
}

// ---------------------------------------------------------------------------
// Fused residual-VQ selection: scores s[m][c] = |c|^2 - 2 r_m . c via WMMA,
// top-8 per word, then greedy distinct argmin per group of 8 words.
// One block = 16 rows (2 quantization groups), 8 waves stream the codebook
// independently (no block barriers in the hot loop: each wave touches only
// its own LDS slice, and per-wave LDS ops are architecturally in-order).
// ---------------------------------------------------------------------------
__device__ inline void top8_insert(float (&tv)[8], int (&ti)[8], float v, int c)
{
    if (v >= tv[7]) return;
    float cv = v; int ci = c;
    #pragma unroll
    for (int s = 0; s < 8; ++s) {
        const bool sw = cv < tv[s];
        const float nv = sw ? cv : tv[s];
        const int   ni = sw ? ci : ti[s];
        cv = sw ? tv[s] : cv;
        ci = sw ? ti[s] : ci;
        tv[s] = nv; ti[s] = ni;
    }
}

__global__ __launch_bounds__(256)
void vq_select(const float* __restrict__ ze, const float* __restrict__ zq,
               const bf16_t* __restrict__ mcb, const float* __restrict__ cbsq,
               int* __restrict__ sel)
{
    __shared__ bf16_t sR[16 * 272];           // residual tile, bf16, padded
    __shared__ float  sScore[8][256];         // per-wave private 16x16 score tile
    __shared__ float  sCandV[16][128];
    __shared__ int    sCandI[16][128];
    __shared__ float  sTopV[16][8];
    __shared__ int    sTopI[16][8];

    const int tid = threadIdx.x, lane = tid & 31, wave = tid >> 5;
    const long long M0 = (long long)blockIdx.x * 16;

    {   // residual = z_e - z_q, converted to bf16 in LDS
        const int row = tid >> 4, c0 = (tid & 15) * 16;
        const float* pe = ze + (M0 + row) * 256 + c0;
        const float* pq = zq + (M0 + row) * 256 + c0;
        bf16_t* d = sR + row * 272 + c0;
        #pragma unroll
        for (int i = 0; i < 16; ++i) d[i] = (bf16_t)(pe[i] - pq[i]);
    }
    __syncthreads();

    const int rl = lane & 15;
    const int h  = lane >> 4;
    v16bf aF[8];
    #pragma unroll
    for (int kk = 0; kk < 8; ++kk)
        aF[kk] = make_afrag(sR + rl * 272 + kk * 32, h);

    float topV[8]; int topI[8];
    #pragma unroll
    for (int s = 0; s < 8; ++s) { topV[s] = 3.0e38f; topI[s] = -1; }
    const int colBase = h * 8;   // this lane scans 8 of the 16 columns for row rl

    const int nTiles = NCODE / 16;                 // 1024
    for (int t = wave; t < nTiles; t += 8) {
        const int n0 = t * 16;
        const bf16_t* bp = mcb + (long long)(n0 + rl) * 256 + h * 16;
        __builtin_prefetch((const void*)(bp + 128 * 256), 0, 0);  // next tile for this wave
        v8f acc = {};
        #pragma unroll
        for (int kk = 0; kk < 8; ++kk) {
            v16bf bF = *(const v16bf*)(bp + kk * 32);
            acc = __builtin_amdgcn_wmma_f32_16x16x32_bf16(false, aF[kk], false, bF, (short)0, acc, false, false);
        }
        const float csq = cbsq[n0 + rl];
        #pragma unroll
        for (int r = 0; r < 8; ++r) {
            const int m = r + (h << 3);
            sScore[wave][m * 16 + rl] = csq - 2.0f * acc[r];
        }
        // Per-wave LDS ops are in-order (DScnt); fence only stops compiler motion.
        asm volatile("" ::: "memory");
        #pragma unroll
        for (int c = 0; c < 8; ++c) {
            const float v = sScore[wave][rl * 16 + colBase + c];
            top8_insert(topV, topI, v, n0 + colBase + c);
        }
        asm volatile("" ::: "memory");
    }

    {   // dump per-lane candidate lists
        const int slot = wave * 16 + h * 8;
        #pragma unroll
        for (int s = 0; s < 8; ++s) { sCandV[rl][slot + s] = topV[s]; sCandI[rl][slot + s] = topI[s]; }
    }
    __syncthreads();

    if (tid < 16) {  // merge 128 candidates -> final top-8 per row
        float tv[8]; int ti[8];
        #pragma unroll
        for (int s = 0; s < 8; ++s) { tv[s] = 3.0e38f; ti[s] = -1; }
        for (int s = 0; s < 128; ++s) top8_insert(tv, ti, sCandV[tid][s], sCandI[tid][s]);
        #pragma unroll
        for (int s = 0; s < 8; ++s) { sTopV[tid][s] = tv[s]; sTopI[tid][s] = ti[s]; }
    }
    __syncthreads();

    if (tid < 2) {   // greedy distinct selection across the 8 words of a group
        int ex[8];
        #pragma unroll
        for (int j = 0; j < 8; ++j) {
            const int row = tid * 8 + j;
            int pick = -1;
            #pragma unroll
            for (int s = 0; s < 8; ++s) {
                const int c = sTopI[row][s];
                bool used = false;
                #pragma unroll
                for (int e = 0; e < 8; ++e)
                    if (e < j) used |= (ex[e] == c);
                if (pick < 0 && !used) pick = c;
            }
            ex[j] = pick;
            sel[M0 + row] = pick;
        }
    }
}

// ---------------------------------------------------------------------------
// Small elementwise / reduction kernels
// ---------------------------------------------------------------------------
__global__ void zero_kernel(float* p) {
    p[(long long)blockIdx.x * 256 + threadIdx.x] = 0.0f;
}
__global__ void copy_kernel(const float* __restrict__ in, float* __restrict__ out) {
    const long long i = (long long)blockIdx.x * 256 + threadIdx.x;
    out[i] = in[i];
}
__global__ void cvt_kernel(const float* __restrict__ in, bf16_t* __restrict__ out) {
    const long long i = (long long)blockIdx.x * 256 + threadIdx.x;
    out[i] = (bf16_t)in[i];
}
__global__ void cbsq_kernel(const float* __restrict__ mcb, float* __restrict__ cbsq) {
    const int v = blockIdx.x * 256 + threadIdx.x;
    const float* p = mcb + (long long)v * 256;
    float s = 0.0f;
    for (int d = 0; d < 256; ++d) s += p[d] * p[d];
    cbsq[v] = s;
}
__global__ void zq_update(float* __restrict__ zq, const float* __restrict__ mcb,
                          const int* __restrict__ sel) {
    const long long i = (long long)blockIdx.x * 256 + threadIdx.x;
    const long long m = i >> 8;
    const int d = (int)(i & 255);
    zq[i] += mcb[(long long)sel[m] * 256 + d];
}
__global__ void tok_kernel(const int* __restrict__ sel, const int* __restrict__ token_id,
                           float* __restrict__ out) {
    const int i = blockIdx.x * 256 + threadIdx.x;
    out[i] = (float)token_id[sel[i]];
}
__global__ __launch_bounds__(256)
void loss_partial(const float* __restrict__ a, const float* __restrict__ b,
                  float* __restrict__ partial) {
    __shared__ float red[256];
    const int t = threadIdx.x;
    const long long base = (long long)blockIdx.x * 4096 + t;
    float s = 0.0f;
    #pragma unroll
    for (int i = 0; i < 16; ++i) {
        const long long idx = base + i * 256;
        const float d = a[idx] - b[idx];
        s += d * d;
    }
    red[t] = s; __syncthreads();
    for (int st = 128; st > 0; st >>= 1) { if (t < st) red[t] += red[t + st]; __syncthreads(); }
    if (t == 0) partial[blockIdx.x] = red[0];
}
__global__ __launch_bounds__(256)
void loss_final(const float* __restrict__ partial, float* __restrict__ out) {
    __shared__ float red[256];
    const int t = threadIdx.x;
    float s = 0.0f;
    for (int i = t; i < 3 * 1024; i += 256) s += partial[i];
    red[t] = s; __syncthreads();
    for (int st = 128; st > 0; st >>= 1) { if (t < st) red[t] += red[t + st]; __syncthreads(); }
    if (t == 0) out[0] = red[0] * (1.0f / (float)(NTOK * WD));
}

// ---------------------------------------------------------------------------
// Host orchestration
// ---------------------------------------------------------------------------
extern "C" void kernel_launch(void* const* d_in, const int* in_sizes, int n_in,
                              void* d_out, int out_size, void* d_ws, size_t ws_size,
                              hipStream_t stream)
{
    (void)in_sizes; (void)n_in; (void)out_size; (void)ws_size;
    auto P = [&](int i) { return (const float*)d_in[i]; };
    // setup_inputs() insertion order:
    // 0 x, 1 lin_w, 2 lin_b, 3 rev_w, 4 rev_b,
    // 5..16 enc {in_w,in_b,out_w,out_b,l1_w,l1_b,l2_w,l2_b,ln1_w,ln1_b,ln2_w,ln2_b},
    // 17..28 dec (same), 29 cb_map_w, 30 cb_map_b, 31 codebook, 32 token_id
    const int* token_id = (const int*)d_in[32];
    float* fout = (float*)d_out;

    // -------- workspace layout (bump allocator, 256B aligned) --------
    char* ws = (char*)d_ws;
    size_t cur = 0;
    auto alloc = [&](size_t bytes) -> void* {
        void* p = ws + cur;
        cur = (cur + bytes + 255) & ~(size_t)255;
        return p;
    };
    float*  mcb    = (float*) alloc((size_t)NCODE * WD * 4);
    bf16_t* mcbb   = (bf16_t*)alloc((size_t)NCODE * WD * 2);
    float*  cbsq   = (float*) alloc((size_t)NCODE * 4);
    float*  linout = (float*) alloc((size_t)NTOK * WD * 4);   // reused as decoder out
    float*  ze     = (float*) alloc((size_t)NTOK * WD * 4);
    float*  zq     = (float*) alloc((size_t)NTOK * WD * 4);
    float*  qkv    = (float*) alloc((size_t)NTOK * 768 * 4);
    float*  attn_a = (float*) alloc((size_t)NTOK * WD * 4);
    float*  attnout= (float*) alloc((size_t)NTOK * WD * 4);
    float*  x1     = (float*) alloc((size_t)NTOK * WD * 4);
    float*  hbuf   = (float*) alloc((size_t)NTOK * 2048 * 4);
    float*  ffout  = (float*) alloc((size_t)NTOK * WD * 4);
    int*    sel    = (int*)   alloc((size_t)NTOK * 4);
    float*  lpart  = (float*) alloc((size_t)3 * 1024 * 4);

    auto gemm = [&](const float* A, int lda, long long aB,
                    const float* W, int ldw, long long wB, int wT,
                    const float* bias, long long bB,
                    float* C, int ldc, long long cB,
                    int M, int N, int K, int relu, int batch) {
        dim3 g(N / 64, M / 128, batch);
        gemm_wmma<<<g, 256, 0, stream>>>(A, lda, aB, W, ldw, wB, wT,
                                         bias, bB, C, ldc, cB, M, N, K, relu);
    };

    auto run_tf = [&](const float* X, int p, float* OUT) {
        const float *in_w = P(p+0), *in_b = P(p+1), *out_w = P(p+2), *out_b = P(p+3),
                    *l1w = P(p+4), *l1b = P(p+5), *l2w = P(p+6), *l2b = P(p+7),
                    *g1 = P(p+8), *b1 = P(p+9), *g2 = P(p+10), *b2 = P(p+11);
        gemm(X, 256, 0, in_w, 256, 0, 0, in_b, 0, qkv, 768, 0, NTOK, 768, 256, 0, 1);
        attn_kernel<<<2048, 256, 0, stream>>>(qkv, attn_a);
        gemm(attn_a, 256, 0, out_w, 256, 0, 0, out_b, 0, attnout, 256, 0, NTOK, 256, 256, 0, 1);
        ln_kernel<<<NTOK, 256, 0, stream>>>(X, attnout, g1, b1, x1);
        gemm(x1, 256, 0, l1w, 256, 0, 0, l1b, 0, hbuf, 2048, 0, NTOK, 2048, 256, 1, 1);
        gemm(hbuf, 2048, 0, l2w, 2048, 0, 0, l2b, 0, ffout, 256, 0, NTOK, 256, 2048, 0, 1);
        ln_kernel<<<NTOK, 256, 0, stream>>>(x1, ffout, g2, b2, OUT);
    };

    // 1. linear_out[b,w,:] = x[b,:] @ lin_w[w]  (+ lin_b[w]) ; lin_w slice is (K=64 x N=256)
    gemm(P(0), 64, 0,
         P(1), 256, 64 * 256, /*wTrans=*/1,
         P(2), 256,
         linout, 2048, 256,
         2048, 256, 64, 0, /*batch=*/8);

    // 2. encoder transformer -> z_e
    run_tf(linout, 5, ze);

    // 3. mapped_codebook = codebook @ cb_map_w.T + cb_map_b
    gemm(P(31), 4096, 0, P(29), 4096, 0, 0, P(30), 0, mcb, 256, 0, NCODE, 256, 4096, 0, 1);
    cvt_kernel<<<NCODE * WD / 256, 256, 0, stream>>>(mcb, mcbb);
    cbsq_kernel<<<NCODE / 256, 256, 0, stream>>>(mcb, cbsq);

    // 4. residual VQ: 3 layers of fused distance+top8+greedy selection
    zero_kernel<<<NTOK * WD / 256, 256, 0, stream>>>(zq);
    for (int L = 0; L < 3; ++L) {
        vq_select<<<NTOK / 16, 256, 0, stream>>>(ze, zq, mcbb, cbsq, sel);
        if (L == 0) {
            tok_kernel<<<NTOK / 256, 256, 0, stream>>>(sel, token_id, fout + TOK_OFF);
            copy_kernel<<<NTOK * WD / 256, 256, 0, stream>>>(ze, fout + TTQ_OFF); // ttq = first residual = z_e
        }
        zq_update<<<NTOK * WD / 256, 256, 0, stream>>>(zq, mcb, sel);
        loss_partial<<<1024, 256, 0, stream>>>(zq, ze, lpart + L * 1024);
    }
    loss_final<<<1, 256, 0, stream>>>(lpart, fout + LOSS_OFF);

    // 5. decoder transformer on z_q (straight-through value == z_q)
    run_tf(zq, 17, linout);

    // 6. x_hat = dec_out.reshape(2048,2048) @ rev_w.T + rev_b  -> d_out[0:131072]
    gemm(linout, 2048, 0, P(3), 2048, 0, 0, P(4), 0, fout, 64, 0, 2048, 64, 2048, 0, 1);
}
